// ShortTermLSTM_41394894799054
// MI455X (gfx1250) — matrix operand order repaired and checked by reading, not compiled, not run on hardware
//
#include <hip/hip_runtime.h>

// ---------------- problem dims ----------------
constexpr int NCELL = 16384;
constexpr int NB    = 3;
constexpr int T     = 128;
constexpr int IN    = 4;
constexpr int H     = 100;

// ---------------- packed / padded dims ----------------
constexpr int Hp    = 112;       // per-gate cols padded to 7*16
constexpr int Gp    = 4 * Hp;    // 448 padded gate columns
constexpr int Kp    = 128;       // K (=H+IN) padded to 4 chunks of 32
constexpr int COLH  = 136;       // halves per padded K-column/row (272 B, bank-skewed)
constexpr int MROWS = 32;        // cells per workgroup (2 row-tiles)
constexpr int NWAVE = 7;         // one wave per 16-unit chunk, owns all 4 gates
constexpr int NTHR  = NWAVE * 32;

// workspace layout (bytes)
constexpr int WS_BSTRIDE_H = Gp * COLH;              // halves per branch = 60928
constexpr int WS_BIAS_OFF  = NB * WS_BSTRIDE_H * 2;  // 365568 bytes

// LDS layout (bytes)
constexpr int L_B    = 0;                  // [448][136] f16 = 121856
constexpr int L_BIAS = 121856;             // [448] f32      = 1792
constexpr int L_A0   = 123648;             // [32][136] f16  = 8704
constexpr int L_A1   = 132352;             // [32][136] f16  = 8704
constexpr int L_TOT  = 141056;

typedef _Float16 v8h  __attribute__((ext_vector_type(8)));
typedef _Float16 v16h __attribute__((ext_vector_type(16)));
typedef float    v8f  __attribute__((ext_vector_type(8)));

__device__ __forceinline__ float fsig(float x) { return 1.f / (1.f + __expf(-x)); }
__device__ __forceinline__ float ftanh(float x) {
    float ax = fabsf(x);
    float e  = __expf(-2.f * ax);
    float r  = (1.f - e) / (1.f + e);
    return copysignf(r, x);
}

// Pack B = [Wh^T ; Wi^T ; 0] per branch, column-major (K contiguous), skewed stride, f16.
__global__ void pack_weights_k(const float* __restrict__ w_ih,
                               const float* __restrict__ w_hh,
                               _Float16* __restrict__ wsB) {
    int idx = blockIdx.x * blockDim.x + threadIdx.x;
    if (idx >= NB * Gp * Kp) return;
    int b   = idx / (Gp * Kp);
    int r   = idx % (Gp * Kp);
    int col = r / Kp;
    int k   = r % Kp;
    int g = col / Hp, u = col % Hp;
    float v = 0.f;
    if (u < H) {
        int j = g * H + u;
        if (k < H)            v = w_hh[(b * 4 * H + j) * H + k];
        else if (k < H + IN)  v = w_ih[(b * 4 * H + j) * IN + (k - H)];
    }
    wsB[b * WS_BSTRIDE_H + col * COLH + k] = (_Float16)v;
}

__global__ void pack_bias_k(const float* __restrict__ b_ih,
                            const float* __restrict__ b_hh,
                            float* __restrict__ wsBias) {
    int idx = blockIdx.x * blockDim.x + threadIdx.x;
    if (idx >= NB * Gp) return;
    int b = idx / Gp, col = idx % Gp;
    int g = col / Hp, u = col % Hp;
    float v = 0.f;
    if (u < H) {
        int j = g * H + u;
        v = b_ih[b * 4 * H + j] + b_hh[b * 4 * H + j];
    }
    wsBias[idx] = v;
}

// One workgroup = 32 cells of one branch. 7 waves; wave uc owns unit columns
// [uc*16, uc*16+16) of ALL FOUR gates -> elementwise update fully in registers.
// K-chunk outermost in the MAC loop; accumulators start at zero and the scalar
// bias is folded into the elementwise phase (no splat-vector spills).
__global__ void lstm_wmma_k(const float* __restrict__ x,
                            const _Float16* __restrict__ wsB,
                            const float* __restrict__ wsBias,
                            float* __restrict__ out) {
    extern __shared__ char smem[];
    _Float16* Bl    = (_Float16*)(smem + L_B);
    float*    BiasL = (float*)(smem + L_BIAS);
    _Float16* Ab0   = (_Float16*)(smem + L_A0);
    _Float16* Ab1   = (_Float16*)(smem + L_A1);

    const int tid  = threadIdx.x;
    const int b    = blockIdx.x / (NCELL / MROWS);
    const int tile = blockIdx.x % (NCELL / MROWS);
    const int uc   = tid >> 5;            // wave id = unit chunk (0..6)
    const int l    = tid & 31;
    const int ln   = l & 15;
    const int hi   = (l >= 16) ? 1 : 0;
    const int u    = uc * 16 + ln;        // padded unit column owned by this lane

    // ---- one-time LDS fill: resident weights + bias, zero both A panels, x(t=0) ----
    {
        const int4* src = (const int4*)(wsB + (size_t)b * WS_BSTRIDE_H);
        int4*       dst = (int4*)Bl;
        for (int i = tid; i < (Gp * COLH * 2) / 16; i += NTHR) dst[i] = src[i];
    }
    for (int i = tid; i < Gp; i += NTHR) BiasL[i] = wsBias[b * Gp + i];
    for (int i = tid; i < (2 * MROWS * COLH) / 2; i += NTHR) ((unsigned int*)Ab0)[i] = 0u;
    if (tid < MROWS * IN) {
        int cell = tid >> 2, ii = tid & 3;
        long n = (long)tile * MROWS + cell;
        float v = x[((n * NB + b) * T + 0) * IN + ii];
        Ab0[cell * COLH + H + ii] = (_Float16)v;
    }
    __syncthreads();

    float bv[4];
#pragma unroll
    for (int g = 0; g < 4; ++g) bv[g] = BiasL[g * Hp + u];

    float cst[2][8];
#pragma unroll
    for (int rt = 0; rt < 2; ++rt)
#pragma unroll
        for (int r = 0; r < 8; ++r) cst[rt][r] = 0.f;

    for (int t = 0; t < T; ++t) {
        _Float16* Ac = (t & 1) ? Ab1 : Ab0;
        _Float16* An = (t & 1) ? Ab0 : Ab1;

        // prefetch x(t+1) early to hide HBM latency behind WMMA
        float xv = 0.f;
        const int cellx = tid >> 2, iix = tid & 3;
        const bool doX = (tid < MROWS * IN) && (t + 1 < T);
        if (doX) {
            long n = (long)tile * MROWS + cellx;
            xv = x[((n * NB + b) * T + (t + 1)) * IN + iix];
        }

        // ---- accumulators: 4 gates x 2 row-tiles, zero-initialized ----
        v8f acc[4][2];
#pragma unroll
        for (int g = 0; g < 4; ++g)
#pragma unroll
            for (int rt = 0; rt < 2; ++rt)
#pragma unroll
                for (int r = 0; r < 8; ++r) acc[g][rt][r] = 0.f;

        // ---- MAC loop: kc outermost so only the current K-chunk's A frags are live ----
#pragma unroll
        for (int kc = 0; kc < 4; ++kc) {
            v16h af[2];
#pragma unroll
            for (int rt = 0; rt < 2; ++rt) {
                const _Float16* ap = Ac + (rt * 16 + ln) * COLH + kc * 32 + (hi ? 8 : 0);
                v8h lo = *(const v8h*)(ap);
                v8h hh = *(const v8h*)(ap + 16);
#pragma unroll
                for (int e = 0; e < 8; ++e) { af[rt][e] = lo[e]; af[rt][8 + e] = hh[e]; }
            }
#pragma unroll
            for (int g = 0; g < 4; ++g) {
                const int cb = g * Hp + uc * 16;
                const _Float16* bp = Bl + (cb + ln) * COLH + kc * 32 + (hi ? 8 : 0);
                v8h lo = *(const v8h*)(bp);
                v8h hh = *(const v8h*)(bp + 16);
                v16h bfr;
#pragma unroll
                for (int e = 0; e < 8; ++e) { bfr[e] = lo[e]; bfr[8 + e] = hh[e]; }
#pragma unroll
                for (int rt = 0; rt < 2; ++rt) {
                    acc[g][rt] = __builtin_amdgcn_wmma_f32_16x16x32_f16(
                        false, af[rt], false, bfr, (short)0, acc[g][rt], false, false);
                }
            }
        }

        // ---- elementwise LSTM update: entirely in registers (bias folded here) ----
        // C/D layout: lane ln(+16*hi): N = u; VGPR r -> M = rt*16 + r + 8*hi
#pragma unroll
        for (int rt = 0; rt < 2; ++rt)
#pragma unroll
            for (int r = 0; r < 8; ++r) {
                float gi = acc[0][rt][r] + bv[0];
                float gf = acc[1][rt][r] + bv[1];
                float gg = acc[2][rt][r] + bv[2];
                float go = acc[3][rt][r] + bv[3];
                float c  = fsig(gf) * cst[rt][r] + fsig(gi) * ftanh(gg);
                cst[rt][r] = c;
                float h = fsig(go) * ftanh(c);
                int m = rt * 16 + r + 8 * hi;
                if (u < H) {
                    An[m * COLH + u] = (_Float16)h;   // feeds next step's A panel
                    if (t == T - 1) {
                        long n = (long)tile * MROWS + m;
                        out[n * (NB * H) + b * H + u] = h;
                    }
                }
            }
        if (doX) An[cellx * COLH + H + iix] = (_Float16)xv;
        __syncthreads();   // single barrier per step (A panels double-buffered)
    }
}

extern "C" void kernel_launch(void* const* d_in, const int* in_sizes, int n_in,
                              void* d_out, int out_size, void* d_ws, size_t ws_size,
                              hipStream_t stream) {
    (void)in_sizes; (void)n_in; (void)out_size; (void)ws_size;
    const float* x    = (const float*)d_in[0];
    const float* w_ih = (const float*)d_in[1];
    const float* w_hh = (const float*)d_in[2];
    const float* b_ih = (const float*)d_in[3];
    const float* b_hh = (const float*)d_in[4];

    _Float16* wsB    = (_Float16*)d_ws;
    float*    wsBias = (float*)((char*)d_ws + WS_BIAS_OFF);

    pack_weights_k<<<(NB * Gp * Kp + 255) / 256, 256, 0, stream>>>(w_ih, w_hh, wsB);
    pack_bias_k<<<(NB * Gp + 255) / 256, 256, 0, stream>>>(b_ih, b_hh, wsBias);

    lstm_wmma_k<<<NB * (NCELL / MROWS), NTHR, L_TOT, stream>>>(x, wsB, wsBias, (float*)d_out);
}
